// InteractionBlock_5119601016935
// MI455X (gfx1250) — compile-verified, matrix operand order checked by reading, not run on hardware
//
#include <hip/hip_runtime.h>
#include <hip/hip_bf16.h>

// ---------------------------------------------------------------------------
// DimeNet InteractionBlock on gfx1250 (MI455X).
// f16 WMMA w/ f32 accumulate; 2x4 register tiling per wave (8 accumulators);
// weights pre-transposed to f16 once, staged per block via
// GLOBAL_LOAD_ASYNC_TO_LDS_B128 with double buffering (ASYNCcnt pipelining).
// ---------------------------------------------------------------------------

typedef _Float16 v4h  __attribute__((ext_vector_type(4)));
typedef _Float16 v8h  __attribute__((ext_vector_type(8)));
typedef _Float16 v16h __attribute__((ext_vector_type(16)));
typedef float    v4f  __attribute__((ext_vector_type(4)));
typedef float    v8f  __attribute__((ext_vector_type(8)));

#define NTHREADS 256   // 8 waves of 32
#define ROWS     128   // rows per block

// Workspace layout (after the [E,128] f32 x_down region), in halves:
#define OFF_KJ    0        // W_kj^T    128x128
#define OFF_RBF2  16384    // W_rbf2^T  128x128
#define OFF_DOWN  32768    // W_down^T  128x128
#define OFF_SBF2  49152    // W_sbf2^T  128x128
#define OFF_UP    65536    // W_up^T    128x128
#define OFF_R1A   81920    // W_r1a^T   128x128
#define OFF_R1B   98304    // W_r1b^T   128x128
#define OFF_SBF1  114688   // W_sbf1^T  128x64 (K=42 zero-padded to 64)

__device__ __forceinline__ float silu_f(float v) {
  return v * __builtin_amdgcn_rcpf(1.0f + __expf(-v));
}

// ---- async global->LDS copy of 16B chunks (ASYNCcnt). bytes % 4096 == 0 ----
// Each thread issues bytes/4096 instructions; per-wave ASYNCcnt += bytes/4096.
__device__ __forceinline__ void async_copy_16(const void* gsrc, void* ldst,
                                              int bytes, int tid) {
  const unsigned lbase = (unsigned)(uintptr_t)ldst;  // LDS byte offset
  const unsigned long long gbase = (unsigned long long)(uintptr_t)gsrc;
  for (int off = tid * 16; off < bytes; off += NTHREADS * 16) {
    const unsigned laddr = lbase + (unsigned)off;
    asm volatile("global_load_async_to_lds_b128 %0, %1, %2"
                 :
                 : "v"(laddr), "v"(off), "s"(gbase)
                 : "memory");
  }
}
#define WAIT_ASYNC_0() asm volatile("s_wait_asynccnt 0x0" ::: "memory")
#define WAIT_ASYNC_8() asm volatile("s_wait_asynccnt 0x8" ::: "memory")

// Load one 16-element f16 fragment (A or B operand) for a 16x16x32 WMMA.
// Lane half h (=lane>>4) supplies K = {k0+8h..+7, k0+16+8h..+7}: two
// contiguous 128-bit LDS loads.
__device__ __forceinline__ v16h lds_frag(const _Float16* __restrict__ row,
                                         int k0, int half) {
  v8h lo = *(const v8h*)(row + k0 + 8 * half);
  v8h hi = *(const v8h*)(row + k0 + 16 + 8 * half);
  return __builtin_shufflevector(lo, hi, 0, 1, 2, 3, 4, 5, 6, 7,
                                 8, 9, 10, 11, 12, 13, 14, 15);
}

// One wave computes a 2x4 grid of 16x16 output tiles:
//   rows [(mtA+i)*16,+16) i=0..1, cols [(nt0+j)*16,+16) j=0..3.
// Per K-block: 2 A-frags + 4 B-frags (12 ds_load_b128) feed 8 WMMAs.
// A: [ROWS][lda] f16 LDS.  WT: [128][ldw] f16 LDS (weight transposed [n][k]).
template <int KB>
__device__ __forceinline__ void wave_gemm(const _Float16* __restrict__ A, int lda,
                                          const _Float16* __restrict__ WT, int ldw,
                                          int mtA, int nt0, int lane, v8f acc[8]) {
  const int half = lane >> 4;
  const int l15  = lane & 15;
  const _Float16* arow0 = A + (mtA * 16 + l15) * lda;
  const _Float16* arow1 = arow0 + 16 * lda;
#pragma unroll
  for (int kb = 0; kb < KB; ++kb) {
    const int k0 = kb * 32;
    v16h a0 = lds_frag(arow0, k0, half);
    v16h a1 = lds_frag(arow1, k0, half);
#pragma unroll
    for (int j = 0; j < 4; ++j) {
      const _Float16* brow = WT + ((nt0 + j) * 16 + l15) * ldw;
      v16h b = lds_frag(brow, k0, half);
      acc[j] = __builtin_amdgcn_wmma_f32_16x16x32_f16(
          false, a0, false, b, (short)0, acc[j], false, false);
      acc[4 + j] = __builtin_amdgcn_wmma_f32_16x16x32_f16(
          false, a1, false, b, (short)0, acc[4 + j], false, false);
    }
  }
}

__device__ __forceinline__ void zero_acc(v8f acc[8]) {
#pragma unroll
  for (int q = 0; q < 8; ++q) acc[q] = v8f(0.f);
}

// C/D layout: VGPR r -> row r + 8*(lane>>4), col = lane&15 within tile.
template <typename F>
__device__ __forceinline__ void tile_foreach(const v8f acc[8], int mtA, int nt0,
                                             int lane, F&& f) {
  const int colB = lane & 15;
  const int rowB = 8 * (lane >> 4);
#pragma unroll
  for (int i = 0; i < 2; ++i) {
    const int rowH = (mtA + i) * 16 + rowB;
#pragma unroll
    for (int j = 0; j < 4; ++j) {
      const int col = (nt0 + j) * 16 + colB;
#pragma unroll
      for (int r = 0; r < 8; ++r) {
        f(rowH + r, col, acc[i * 4 + j][r]);
      }
    }
  }
}

// Cooperative: [ROWS x 128] f32 tile from global -> f16 LDS, 4-wide
// (v_cvt_pk_f16_f32 + ds_store_b64), zero-pad rows beyond rowsValid.
__device__ __forceinline__ void load_rows(const float* __restrict__ g,
                                          int rowsValid,
                                          _Float16* __restrict__ dst, int tid) {
  for (int idx = tid; idx < ROWS * 32; idx += NTHREADS) {
    const int r = idx >> 5, c4 = (idx & 31) * 4;
    v4f v = v4f(0.f);
    if (r < rowsValid) v = *(const v4f*)(g + (size_t)r * 128 + c4);
    v4h h;
#pragma unroll
    for (int i = 0; i < 4; ++i) h[i] = (_Float16)v[i];
    *(v4h*)&dst[r * 128 + c4] = h;
  }
}

// ---------------------------------------------------------------------------
// Kernel 0: one-shot weight prep.  f32 [K][N] row-major -> f16 transposed
// [N][kPad] into workspace (zero-pad K to kPad).  One block per matrix.
// ---------------------------------------------------------------------------
__global__ __launch_bounds__(NTHREADS) void k_prep_weights(
    const float* __restrict__ W_kj, const float* __restrict__ W_rbf2,
    const float* __restrict__ W_down, const float* __restrict__ W_sbf2,
    const float* __restrict__ W_up, const float* __restrict__ W_r1a,
    const float* __restrict__ W_r1b, const float* __restrict__ W_sbf1,
    _Float16* __restrict__ wbuf) {
  const float* src;
  _Float16* dst;
  int K = 128, N = 128, kPad = 128;
  switch (blockIdx.x) {
    case 0: src = W_kj;   dst = wbuf + OFF_KJ;   break;
    case 1: src = W_rbf2; dst = wbuf + OFF_RBF2; break;
    case 2: src = W_down; dst = wbuf + OFF_DOWN; break;
    case 3: src = W_sbf2; dst = wbuf + OFF_SBF2; break;
    case 4: src = W_up;   dst = wbuf + OFF_UP;   break;
    case 5: src = W_r1a;  dst = wbuf + OFF_R1A;  break;
    case 6: src = W_r1b;  dst = wbuf + OFF_R1B;  break;
    default: src = W_sbf1; dst = wbuf + OFF_SBF1; K = 42; kPad = 64; break;
  }
  for (int idx = threadIdx.x; idx < N * kPad; idx += NTHREADS) {
    const int n = idx / kPad, k = idx - n * kPad;
    dst[idx] = (_Float16)((k < K) ? src[(size_t)k * N + n] : 0.0f);
  }
}

// ---------------------------------------------------------------------------
// Kernel 1: per-edge.  x_down = silu( (silu(x@Wkj+b) * silu(rbf@Wrbf2+b))
//                                     @ Wdown + b )            [E,128] -> ws
// ---------------------------------------------------------------------------
__global__ __launch_bounds__(NTHREADS) void k_edge_down(
    const float* __restrict__ x, const float* __restrict__ rbf,
    const float* __restrict__ b_kj, const float* __restrict__ b_rbf2,
    const float* __restrict__ b_down, const _Float16* __restrict__ wbuf,
    float* __restrict__ x_down, int E) {
  __shared__ __align__(16) _Float16 sW0[128 * 128];
  __shared__ __align__(16) _Float16 sW1[128 * 128];
  __shared__ __align__(16) _Float16 sA[ROWS * 128];
  __shared__ __align__(16) _Float16 sT[ROWS * 128];

  const int tid   = threadIdx.x;
  const int lane  = tid & 31;
  const int wid   = tid >> 5;
  const int mtA   = (wid >> 1) * 2;   // 0,2,4,6
  const int nt0   = (wid & 1) * 4;    // 0 or 4
  const int ebase = blockIdx.x * ROWS;
  const int rowsValid = (E - ebase < ROWS) ? (E - ebase) : ROWS;

  // Prefetch first two weights via async DMA (8 + 8 chunks per thread).
  async_copy_16(wbuf + OFF_KJ,   sW0, 32768, tid);
  async_copy_16(wbuf + OFF_RBF2, sW1, 32768, tid);
  load_rows(x + (size_t)ebase * 128, rowsValid, sA, tid);

  // ---- stage 1: T = silu(x @ W_kj + b_kj) ----
  WAIT_ASYNC_8();  // oldest 8 done => W_kj staged
  __syncthreads();
  v8f acc[8];
  zero_acc(acc);
  wave_gemm<4>(sA, 128, sW0, 128, mtA, nt0, lane, acc);
  __syncthreads();
  tile_foreach(acc, mtA, nt0, lane, [&](int row, int col, float v) {
    sT[row * 128 + col] = (_Float16)silu_f(v + b_kj[col]);
  });
  async_copy_16(wbuf + OFF_DOWN, sW0, 32768, tid);  // prefetch stage-3 weight
  load_rows(rbf + (size_t)ebase * 128, rowsValid, sA, tid);

  // ---- stage 2: T *= silu(rbf @ W_rbf2 + b_rbf2) ----
  WAIT_ASYNC_8();  // oldest 16 done => W_rbf2 staged
  __syncthreads();
  zero_acc(acc);
  wave_gemm<4>(sA, 128, sW1, 128, mtA, nt0, lane, acc);
  tile_foreach(acc, mtA, nt0, lane, [&](int row, int col, float v) {
    const float t = (float)sT[row * 128 + col];
    sT[row * 128 + col] = (_Float16)(silu_f(v + b_rbf2[col]) * t);
  });

  // ---- stage 3: x_down = silu(T @ W_down + b_down) ----
  WAIT_ASYNC_0();
  __syncthreads();
  zero_acc(acc);
  wave_gemm<4>(sT, 128, sW0, 128, mtA, nt0, lane, acc);
  tile_foreach(acc, mtA, nt0, lane, [&](int row, int col, float v) {
    if (row < rowsValid)
      x_down[(size_t)(ebase + row) * 128 + col] = silu_f(v + b_down[col]);
  });
}

// ---------------------------------------------------------------------------
// Kernel 2: per-triplet.  s = silu(silu(sbf@Wsbf1)@Wsbf2);
//   t = x_down[idx_kj] * s;  out = silu(t@Wup + b_up);
//   atomic scatter-add into m[idx_ji].
// ---------------------------------------------------------------------------
__global__ __launch_bounds__(NTHREADS) void k_triplet(
    const float* __restrict__ sbf, const int* __restrict__ idx_kj,
    const int* __restrict__ idx_ji, const float* __restrict__ b_up,
    const _Float16* __restrict__ wbuf, const float* __restrict__ x_down,
    float* __restrict__ m, int T) {
  __shared__ __align__(16) _Float16 sW0[128 * 128];
  __shared__ __align__(16) _Float16 sW1[128 * 128];
  __shared__ __align__(16) _Float16 sA[ROWS * 128];
  __shared__ __align__(16) _Float16 sB[ROWS * 128];

  const int tid   = threadIdx.x;
  const int lane  = tid & 31;
  const int wid   = tid >> 5;
  const int mtA   = (wid >> 1) * 2;
  const int nt0   = (wid & 1) * 4;
  const int tbase = blockIdx.x * ROWS;
  const int rowsValid = (T - tbase < ROWS) ? (T - tbase) : ROWS;

  async_copy_16(wbuf + OFF_SBF1, sW0, 16384, tid);  // 4 chunks
  async_copy_16(wbuf + OFF_SBF2, sW1, 32768, tid);  // 8 chunks (12 total)
  // sbf tile as [ROWS][64], K=42 zero-padded.
  for (int idx = tid; idx < ROWS * 64; idx += NTHREADS) {
    const int r = idx >> 6, c = idx & 63;
    const float v = (r < rowsValid && c < 42)
                        ? sbf[(size_t)(tbase + r) * 42 + c] : 0.0f;
    sA[idx] = (_Float16)v;
  }

  // ---- stage 1: sB = silu(sbfPad @ W_sbf1) ----
  WAIT_ASYNC_8();  // oldest 4 done => W_sbf1 staged
  __syncthreads();
  v8f acc[8];
  zero_acc(acc);
  wave_gemm<2>(sA, 64, sW0, 64, mtA, nt0, lane, acc);
  __syncthreads();
  tile_foreach(acc, mtA, nt0, lane, [&](int row, int col, float v) {
    sB[row * 128 + col] = (_Float16)silu_f(v);
  });
  async_copy_16(wbuf + OFF_UP, sW0, 32768, tid);  // prefetch stage-4 weight

  // ---- stage 2: sA = s = silu(sB @ W_sbf2) ----
  WAIT_ASYNC_8();  // oldest 12 done => W_sbf2 staged
  __syncthreads();
  zero_acc(acc);
  wave_gemm<4>(sB, 128, sW1, 128, mtA, nt0, lane, acc);
  tile_foreach(acc, mtA, nt0, lane, [&](int row, int col, float v) {
    sA[row * 128 + col] = (_Float16)silu_f(v);
  });
  __syncthreads();  // sA read cross-thread below

  // ---- stage 3: sB = f16( x_down[idx_kj[row]] * s )  (gather hits L2) ----
  for (int idx = tid; idx < ROWS * 32; idx += NTHREADS) {
    const int r = idx >> 5, c4 = (idx & 31) * 4;
    v4f g = v4f(0.f);
    if (r < rowsValid) {
      const int e = idx_kj[tbase + r];
      g = *(const v4f*)(x_down + (size_t)e * 128 + c4);
    }
    const v4h s = *(const v4h*)&sA[r * 128 + c4];
    v4h o;
#pragma unroll
    for (int i = 0; i < 4; ++i) o[i] = (_Float16)(g[i] * (float)s[i]);
    *(v4h*)&sB[r * 128 + c4] = o;
  }

  // ---- stage 4: silu(sB @ W_up + b_up), scatter-add into m[idx_ji] ----
  WAIT_ASYNC_0();
  __syncthreads();
  zero_acc(acc);
  wave_gemm<4>(sB, 128, sW0, 128, mtA, nt0, lane, acc);
  tile_foreach(acc, mtA, nt0, lane, [&](int row, int col, float v) {
    if (row < rowsValid) {
      const int e = idx_ji[tbase + row];
      const float val = silu_f(v + b_up[col]);
      __hip_atomic_fetch_add(&m[(size_t)e * 128 + col], val,
                             __ATOMIC_RELAXED, __HIP_MEMORY_SCOPE_AGENT);
    }
  });
}

// ---------------------------------------------------------------------------
// Kernel 3: per-edge residual, in-place on m (= d_out):
//   out = m + silu(silu(m@W_r1a+b)@W_r1b+b) + x
// ---------------------------------------------------------------------------
__global__ __launch_bounds__(NTHREADS) void k_residual(
    const float* __restrict__ x, const float* __restrict__ b_r1a,
    const float* __restrict__ b_r1b, const _Float16* __restrict__ wbuf,
    float* __restrict__ m, int E) {
  __shared__ __align__(16) _Float16 sW0[128 * 128];
  __shared__ __align__(16) _Float16 sW1[128 * 128];
  __shared__ __align__(16) _Float16 sA[ROWS * 128];
  __shared__ __align__(16) _Float16 sB[ROWS * 128];

  const int tid   = threadIdx.x;
  const int lane  = tid & 31;
  const int wid   = tid >> 5;
  const int mtA   = (wid >> 1) * 2;
  const int nt0   = (wid & 1) * 4;
  const int ebase = blockIdx.x * ROWS;
  const int rowsValid = (E - ebase < ROWS) ? (E - ebase) : ROWS;

  async_copy_16(wbuf + OFF_R1A, sW0, 32768, tid);
  async_copy_16(wbuf + OFF_R1B, sW1, 32768, tid);
  load_rows(m + (size_t)ebase * 128, rowsValid, sA, tid);

  // ---- stage 1: sB = silu(m @ W_r1a + b_r1a) ----
  WAIT_ASYNC_8();
  __syncthreads();
  v8f acc[8];
  zero_acc(acc);
  wave_gemm<4>(sA, 128, sW0, 128, mtA, nt0, lane, acc);
  __syncthreads();
  tile_foreach(acc, mtA, nt0, lane, [&](int row, int col, float v) {
    sB[row * 128 + col] = (_Float16)silu_f(v + b_r1a[col]);
  });

  // ---- stage 2: out = m + silu(sB @ W_r1b + b_r1b) + x ----
  WAIT_ASYNC_0();
  __syncthreads();
  zero_acc(acc);
  wave_gemm<4>(sB, 128, sW1, 128, mtA, nt0, lane, acc);
  tile_foreach(acc, mtA, nt0, lane, [&](int row, int col, float v) {
    if (row < rowsValid) {
      const size_t g = (size_t)(ebase + row) * 128 + col;
      m[g] = m[g] + silu_f(v + b_r1b[col]) + x[g];
    }
  });
}

// ---------------------------------------------------------------------------
extern "C" void kernel_launch(void* const* d_in, const int* in_sizes, int n_in,
                              void* d_out, int out_size, void* d_ws,
                              size_t ws_size, hipStream_t stream) {
  (void)n_in; (void)out_size; (void)ws_size;
  const float* x      = (const float*)d_in[0];
  const float* rbf    = (const float*)d_in[1];
  const float* sbf    = (const float*)d_in[2];
  const int*   idx_kj = (const int*)d_in[3];
  const int*   idx_ji = (const int*)d_in[4];
  const float* W_rbf2 = (const float*)d_in[5];
  const float* b_rbf2 = (const float*)d_in[6];
  const float* W_sbf1 = (const float*)d_in[7];
  const float* W_sbf2 = (const float*)d_in[8];
  const float* W_kj   = (const float*)d_in[9];
  const float* b_kj   = (const float*)d_in[10];
  const float* W_down = (const float*)d_in[11];
  const float* b_down = (const float*)d_in[12];
  const float* W_up   = (const float*)d_in[13];
  const float* b_up   = (const float*)d_in[14];
  const float* W_r1a  = (const float*)d_in[15];
  const float* b_r1a  = (const float*)d_in[16];
  const float* W_r1b  = (const float*)d_in[17];
  const float* b_r1b  = (const float*)d_in[18];

  float* out    = (float*)d_out;
  const int E = in_sizes[0] / 128;
  const int T = in_sizes[2] / 42;

  float*     x_down = (float*)d_ws;  // [E,128] f32 (128 MB, L2-resident)
  _Float16*  wbuf   = (_Float16*)((char*)d_ws + (size_t)E * 128 * sizeof(float));

  // Stage weights (f16, transposed) once; zero m (= d_out) for scatter-add.
  k_prep_weights<<<dim3(8), dim3(NTHREADS), 0, stream>>>(
      W_kj, W_rbf2, W_down, W_sbf2, W_up, W_r1a, W_r1b, W_sbf1, wbuf);
  hipMemsetAsync(d_out, 0, (size_t)E * 128 * sizeof(float), stream);

  const dim3 blk(NTHREADS);
  k_edge_down<<<dim3((E + ROWS - 1) / ROWS), blk, 0, stream>>>(
      x, rbf, b_kj, b_rbf2, b_down, wbuf, x_down, E);
  k_triplet<<<dim3((T + ROWS - 1) / ROWS), blk, 0, stream>>>(
      sbf, idx_kj, idx_ji, b_up, wbuf, x_down, out, T);
  k_residual<<<dim3((E + ROWS - 1) / ROWS), blk, 0, stream>>>(
      x, b_r1a, b_r1b, wbuf, out, E);
}
